// HybridDrapeModel_16853451670015
// MI455X (gfx1250) — compile-verified, matrix operand order checked by reading, not run on hardware
//
#include <hip/hip_runtime.h>
#include <hip/hip_bf16.h>

typedef __attribute__((ext_vector_type(16))) __bf16 v16bf;
typedef __attribute__((ext_vector_type(8)))  float  v8f;

#define LAT 128
#define TPB 256   // 8 wave32

__device__ __forceinline__ unsigned short f32_to_bf16(float f) {
    unsigned int u = __float_as_uint(f);
    unsigned int r = (u + 0x7FFFu + ((u >> 16) & 1u)) >> 16;   // round-nearest-even
    return (unsigned short)r;
}
__device__ __forceinline__ unsigned int pack_bf16(float a, float b) {
    return (unsigned int)f32_to_bf16(a) | ((unsigned int)f32_to_bf16(b) << 16);
}

struct MLPDev {
    const unsigned int *w1, *w2, *w3;     // packed bf16 WMMA-B fragments in ws
    const float *b1, *g1, *be1;
    const float *b2, *g2, *be2;
    const float *b3;
};

// ---- pre-shuffle fp32 weights [K][128] into WMMA B-fragment layout, bf16-pair packed ----
// dword t = ((tk*8 + tn)*32 + lane)*8 + i ; lane holds col n=tn*16+(lane&15),
// k = tk*32 + (i<4?0:16) + (lane>>4)*8 + (i&3)*2  (pair k, k+1)
__global__ __launch_bounds__(TPB) void k_prep_w(const float* __restrict__ w, int K, int Kp,
                                                unsigned int* __restrict__ out) {
    int t = blockIdx.x * TPB + threadIdx.x;
    int total = (Kp / 32) * 8 * 256;
    if (t >= total) return;
    int tk   = t / (8 * 256);
    int rem  = t % (8 * 256);
    int tn   = rem / 256;
    int lane = (rem % 256) / 8;
    int i    = rem % 8;
    int n    = tn * 16 + (lane & 15);
    int half = lane >> 4;
    int k    = tk * 32 + (i < 4 ? 0 : 16) + half * 8 + (i & 3) * 2;
    float v0 = (k     < K) ? w[(size_t)k * LAT + n]       : 0.f;
    float v1 = (k + 1 < K) ? w[(size_t)(k + 1) * LAT + n] : 0.f;
    out[t] = pack_bf16(v0, v1);
}

// ---- one GEMM layer: LDS bf16 A-tile [16][KP] x packed B -> v8f acc (16x16 tile per wave) ----
__device__ __forceinline__ v8f wmma_layer(const unsigned short* __restrict__ sA, int KP,
                                          const unsigned int* __restrict__ W,
                                          int wave, int lane) {
    v8f acc{};
    int m = lane & 15, half = lane >> 4;
    for (int tk = 0; tk < KP / 32; ++tk) {
        union { v16bf v; unsigned int u[8]; } A;
#pragma unroll
        for (int i = 0; i < 8; ++i) {
            int k = tk * 32 + (i < 4 ? 0 : 16) + half * 8 + (i & 3) * 2;
            A.u[i] = *(const unsigned int*)&sA[m * KP + k];   // aligned pair (k,k+1)
        }
        union { v16bf v; unsigned int u[8]; } B;
        const unsigned int* bp = W + ((size_t)(tk * 8 + wave) * 32 + lane) * 8;
#pragma unroll
        for (int i = 0; i < 8; ++i) B.u[i] = bp[i];           // 2x global_load_b128
        acc = __builtin_amdgcn_wmma_f32_16x16x32_bf16(false, A.v, false, B.v,
                                                      (short)0, acc, false, false);
    }
    return acc;
}

__device__ __forceinline__ void store_tile_bias(float* __restrict__ sH, v8f acc,
                                                const float* __restrict__ bias,
                                                int wave, int lane) {
    int n  = wave * 16 + (lane & 15);
    int mb = (lane >> 4) * 8;
    float b = bias[n];
#pragma unroll
    for (int i = 0; i < 8; ++i) sH[(mb + i) * LAT + n] = acc[i] + b;
}

// ---- LayerNorm+ReLU, two-stage parallel reduction, bf16-pair packed writeback ----
__device__ __forceinline__ void ln_relu_to_A(const float* __restrict__ sH,
                                             unsigned short* __restrict__ sA,
                                             const float* __restrict__ g,
                                             const float* __restrict__ be,
                                             float* sP, float* sQ,
                                             float* sMu, float* sRs, int tid) {
    __syncthreads();                       // sH complete, prior sA reads complete
    if (tid < 128) {                       // 8 partial sums per row
        int r = tid >> 3, sub = tid & 7;
        const float* p = sH + r * LAT + sub * 16;
        float s = 0.f, s2 = 0.f;
#pragma unroll
        for (int c = 0; c < 16; ++c) { float v = p[c]; s += v; s2 += v * v; }
        sP[tid] = s; sQ[tid] = s2;
    }
    __syncthreads();
    if (tid < 16) {
        float s = 0.f, s2 = 0.f;
#pragma unroll
        for (int j = 0; j < 8; ++j) { s += sP[tid * 8 + j]; s2 += sQ[tid * 8 + j]; }
        float mu = s * (1.f / LAT);
        sMu[tid] = mu;
        sRs[tid] = rsqrtf(s2 * (1.f / LAT) - mu * mu + 1e-5f);
    }
    __syncthreads();
    for (int idx = tid; idx < 16 * 64; idx += TPB) {   // bf16 pairs
        int r = idx >> 6, c = (idx & 63) * 2;
        float mu = sMu[r], rs = sRs[r];
        float v0 = (sH[r * LAT + c]     - mu) * rs * g[c]     + be[c];
        float v1 = (sH[r * LAT + c + 1] - mu) * rs * g[c + 1] + be[c + 1];
        *(unsigned int*)&sA[r * LAT + c] =
            pack_bf16(v0 > 0.f ? v0 : 0.f, v1 > 0.f ? v1 : 0.f);
    }
    __syncthreads();
}

// ---- fused 3-layer MLP on a 16-row tile; returns final 16x16 f32 tile (bias added) ----
__device__ __forceinline__ v8f mlp3(unsigned short* sA, float* sH, float* sP, float* sQ,
                                    float* sMu, float* sRs,
                                    const MLPDev& p, int KP1, int tid) {
    int wave = tid >> 5, lane = tid & 31;
    __syncthreads();                                   // sA tile built
    v8f a = wmma_layer(sA, KP1, p.w1, wave, lane);
    store_tile_bias(sH, a, p.b1, wave, lane);
    ln_relu_to_A(sH, sA, p.g1, p.be1, sP, sQ, sMu, sRs, tid);
    a = wmma_layer(sA, LAT, p.w2, wave, lane);
    store_tile_bias(sH, a, p.b2, wave, lane);
    ln_relu_to_A(sH, sA, p.g2, p.be2, sP, sQ, sMu, sRs, tid);
    a = wmma_layer(sA, LAT, p.w3, wave, lane);
    float b = p.b3[wave * 16 + (lane & 15)];
#pragma unroll
    for (int i = 0; i < 8; ++i) a[i] += b;
    return a;
}

#define MLP_SHARED \
    __shared__ unsigned short sA[16 * 384]; \
    __shared__ float sH[16 * LAT]; \
    __shared__ float sP[128], sQ[128]; \
    __shared__ float sMu[16], sRs[16];

// ---- style projection head: [8,384]@[384,256] -> gelu -> @[256,128] (tiny, 1 block) ----
__global__ __launch_bounds__(TPB) void k_style(const float* __restrict__ feat,
                                               const float* __restrict__ w1, const float* __restrict__ b1,
                                               const float* __restrict__ w2, const float* __restrict__ b2,
                                               float* __restrict__ style) {
    __shared__ float sG[8 * 256];
    int t = threadIdx.x;
    {
        int n = t;
        for (int r = 0; r < 8; ++r) {
            float s = b1[n];
            for (int k = 0; k < 384; ++k) s += feat[r * 384 + k] * w1[k * 256 + n];
            float u = 0.7978845608028654f * (s + 0.044715f * s * s * s);   // gelu(tanh)
            sG[r * 256 + n] = 0.5f * s * (1.f + tanhf(u));
        }
    }
    __syncthreads();
    if (t < 128) {
        for (int r = 0; r < 8; ++r) {
            float s = b2[t];
            for (int k = 0; k < 256; ++k) s += sG[r * 256 + k] * w2[k * 128 + t];
            style[r * 128 + t] = s;
        }
    }
}

// ---- node encoder: concat(pos, style[batch], smpl[batch], mat[batch]) -> MLP(160p->128) ----
__global__ __launch_bounds__(TPB) void k_node_encode(const float* __restrict__ pos,
                                                     const float* __restrict__ style,
                                                     const float* __restrict__ smpl,
                                                     const float* __restrict__ mat,
                                                     const int* __restrict__ batch,
                                                     MLPDev p, float* __restrict__ x,
                                                     unsigned short* __restrict__ xb, int N) {
    MLP_SHARED
    int tid = threadIdx.x;
    int base = blockIdx.x * 16;
    const int KP = 160;   // 145 padded to 5 k-tiles
    for (int idx = tid; idx < 16 * KP; idx += TPB) {
        int r = idx / KP, c = idx % KP;
        int node = base + r; if (node >= N) node = N - 1;
        float v;
        if (c < 3)        v = pos[node * 3 + c];
        else if (c < 131) { int g = batch[node]; v = style[g * 128 + (c - 3)]; }
        else if (c < 141) { int g = batch[node]; v = smpl[g * 10 + (c - 131)]; }
        else if (c < 145) { int g = batch[node]; v = mat[g * 4 + (c - 141)]; }
        else              v = 0.f;
        sA[idx] = f32_to_bf16(v);
    }
    v8f acc = mlp3(sA, sH, sP, sQ, sMu, sRs, p, KP, tid);
    int wave = tid >> 5, lane = tid & 31;
    int n = wave * 16 + (lane & 15), mb = (lane >> 4) * 8;
#pragma unroll
    for (int i = 0; i < 8; ++i) {
        int node = base + mb + i;
        if (node < N) {
            x[(size_t)node * LAT + n]  = acc[i];
            xb[(size_t)node * LAT + n] = f32_to_bf16(acc[i]);   // bf16 mirror for gathers
        }
    }
}

// ---- edge encoder: edge_attr[E,4] (pad 32) -> MLP -> ea ----
__global__ __launch_bounds__(TPB) void k_edge_encode(const float* __restrict__ eattr,
                                                     MLPDev p, float* __restrict__ ea, int E) {
    MLP_SHARED
    int tid = threadIdx.x;
    int base = blockIdx.x * 16;
    const int KP = 32;
    for (int idx = tid; idx < 16 * KP; idx += TPB) {
        int r = idx / KP, c = idx % KP;
        int e = base + r; if (e >= E) e = E - 1;
        sA[idx] = f32_to_bf16(c < 4 ? eattr[(size_t)e * 4 + c] : 0.f);
    }
    v8f acc = mlp3(sA, sH, sP, sQ, sMu, sRs, p, KP, tid);
    int wave = tid >> 5, lane = tid & 31;
    int n = wave * 16 + (lane & 15), mb = (lane >> 4) * 8;
#pragma unroll
    for (int i = 0; i < 8; ++i)
        if (base + mb + i < E) ea[(size_t)(base + mb + i) * LAT + n] = acc[i];
}

// ---- per-block edge MLP: async bf16 gathers of x[row],x[col] -> LDS, ea -> bf16,
//      MLP(384->128), ea += out (f32), scatter-add msgs ----
__global__ __launch_bounds__(TPB) void k_edge_block(const unsigned short* __restrict__ xb,
                                                    const int* __restrict__ ei, int E,
                                                    MLPDev p, float* __restrict__ ea,
                                                    float* __restrict__ msgs) {
    MLP_SHARED
    __shared__ int sRow[16], sCol[16];
    int tid = threadIdx.x;
    int base = blockIdx.x * 16;
    if (tid < 16) {
        int e = base + tid; if (e >= E) e = E - 1;
        sRow[tid] = ei[e];
        sCol[tid] = ei[E + e];
    }
    __syncthreads();
    // async-gather: 32 bf16 rows (x[row], x[col]) x 256B = 512 x b128 transfers
    unsigned int saBase = (unsigned int)(unsigned long long)(void*)sA;
    for (int t = tid; t < 512; t += TPB) {
        int chunk = t & 15;                 // 16-byte chunk within a 256B row
        int seg   = t >> 4;                 // 0..31 : [src(1) | r(4)]
        int r     = seg & 15;
        int src   = seg >> 4;               // 0 = x[row], 1 = x[col]
        int node  = src ? sCol[r] : sRow[r];
        unsigned long long ga =
            (unsigned long long)(const void*)(xb + (size_t)node * LAT + chunk * 8);
        unsigned int la = saBase + (unsigned int)((r * 384 + src * 128 + chunk * 8) * 2);
        asm volatile("global_load_async_to_lds_b128 %0, %1, off"
                     :: "v"(la), "v"(ga) : "memory");
    }
    // ea (f32) -> bf16 pairs into cols [256,384)
    for (int idx = tid; idx < 16 * 64; idx += TPB) {
        int r = idx >> 6, c = (idx & 63) * 2;
        int e = base + r; if (e >= E) e = E - 1;
        *(unsigned int*)&sA[r * 384 + 256 + c] =
            pack_bf16(ea[(size_t)e * LAT + c], ea[(size_t)e * LAT + c + 1]);
    }
    asm volatile("s_wait_asynccnt 0x0" ::: "memory");   // LDS writes landed (this wave)
    v8f acc = mlp3(sA, sH, sP, sQ, sMu, sRs, p, 384, tid);  // barrier inside covers all waves
    int wave = tid >> 5, lane = tid & 31;
    int n = wave * 16 + (lane & 15), mb = (lane >> 4) * 8;
#pragma unroll
    for (int i = 0; i < 8; ++i) {
        int e = base + mb + i;
        if (e < E) {
            float v = ea[(size_t)e * LAT + n] + acc[i];     // residual in f32
            ea[(size_t)e * LAT + n] = v;
            atomicAdd(&msgs[(size_t)sCol[mb + i] * LAT + n], v);   // segment_sum
        }
    }
}

// ---- per-block node MLP: concat(x_bf16 raw copy, msgs->bf16) -> MLP(256->128), x += out ----
__global__ __launch_bounds__(TPB) void k_node_block(const float* __restrict__ msgs,
                                                    MLPDev p, float* __restrict__ x,
                                                    unsigned short* __restrict__ xb, int N) {
    MLP_SHARED
    int tid = threadIdx.x;
    int base = blockIdx.x * 16;
    const unsigned int* xbd = (const unsigned int*)xb;
    for (int idx = tid; idx < 16 * 128; idx += TPB) {   // dwords: 64 x-pairs + 64 msg-pairs
        int r = idx >> 7, d = idx & 127;
        int node = base + r; if (node >= N) node = N - 1;
        unsigned int pk;
        if (d < 64) pk = xbd[(size_t)node * 64 + d];    // raw bf16 copy, no conversion
        else {
            int c = (d - 64) * 2;
            pk = pack_bf16(msgs[(size_t)node * LAT + c], msgs[(size_t)node * LAT + c + 1]);
        }
        *(unsigned int*)&sA[r * 256 + d * 2] = pk;
    }
    v8f acc = mlp3(sA, sH, sP, sQ, sMu, sRs, p, 256, tid);
    int wave = tid >> 5, lane = tid & 31;
    int n = wave * 16 + (lane & 15), mb = (lane >> 4) * 8;
#pragma unroll
    for (int i = 0; i < 8; ++i) {
        int node = base + mb + i;
        if (node < N) {
            float v = x[(size_t)node * LAT + n] + acc[i];
            x[(size_t)node * LAT + n]  = v;
            xb[(size_t)node * LAT + n] = f32_to_bf16(v);
        }
    }
}

// ---- decoder: x @ [128,3] + b ----
__global__ __launch_bounds__(TPB) void k_decode(const float* __restrict__ x,
                                                const float* __restrict__ w,
                                                const float* __restrict__ b,
                                                float* __restrict__ out, int N) {
    int i = blockIdx.x * TPB + threadIdx.x;
    if (i >= N) return;
    float s0 = b[0], s1 = b[1], s2 = b[2];
    for (int k = 0; k < 128; ++k) {
        float v = x[(size_t)i * LAT + k];
        s0 += v * w[k * 3 + 0]; s1 += v * w[k * 3 + 1]; s2 += v * w[k * 3 + 2];
    }
    out[i * 3 + 0] = s0; out[i * 3 + 1] = s1; out[i * 3 + 2] = s2;
}

// ============================ host side ============================
extern "C" void kernel_launch(void* const* d_in, const int* in_sizes, int n_in,
                              void* d_out, int out_size, void* d_ws, size_t ws_size,
                              hipStream_t stream) {
    const int N = in_sizes[0] / 3;
    const int E = in_sizes[4] / 4;

    const float* pos       = (const float*)d_in[0];
    const float* features  = (const float*)d_in[1];
    const float* smpl      = (const float*)d_in[2];
    const float* mat       = (const float*)d_in[3];
    const float* edge_attr = (const float*)d_in[4];

    // params pytree (alphabetical dict flatten): blocks(10x[edge_mlp,node_mlp] x 10 leaves),
    // dec(b,w), edge_enc(10), node_enc(10), proj1(b,w), proj2(b,w)
    // mlp leaf order: be1,be2,g1,g2,l1.b,l1.w,l2.b,l2.w,l3.b,l3.w
    const int PB = 5;
    const int IDX_DEC_B    = PB + 200, IDX_DEC_W = PB + 201;
    const int IDX_EDGE_ENC = PB + 202;
    const int IDX_NODE_ENC = PB + 212;
    const int IDX_P1B = PB + 222, IDX_P1W = PB + 223;
    const int IDX_P2B = PB + 224, IDX_P2W = PB + 225;
    const int* edge_index = (const int*)d_in[PB + 226];
    const int* batch      = (const int*)d_in[PB + 227];

    // workspace layout
    char* ws = (char*)d_ws;
    size_t off = 0;
    float* x    = (float*)(ws + off);          off += (size_t)N * LAT * 4;
    float* msgs = (float*)(ws + off);          off += (size_t)N * LAT * 4;
    float* ea   = (float*)(ws + off);          off += (size_t)E * LAT * 4;
    unsigned short* xb = (unsigned short*)(ws + off); off += (size_t)N * LAT * 2;
    float* style = (float*)(ws + off);         off += 8 * 128 * 4;

    auto prep = [&](int widx, int K, int Kp) -> const unsigned int* {
        unsigned int* dst = (unsigned int*)(ws + off);
        int total = (Kp / 32) * 8 * 256;
        off += (size_t)total * 4;
        k_prep_w<<<(total + TPB - 1) / TPB, TPB, 0, stream>>>((const float*)d_in[widx], K, Kp, dst);
        return dst;
    };
    auto mkmlp = [&](int b, int K1, int Kp1) -> MLPDev {
        MLPDev m;
        m.be1 = (const float*)d_in[b + 0]; m.be2 = (const float*)d_in[b + 1];
        m.g1  = (const float*)d_in[b + 2]; m.g2  = (const float*)d_in[b + 3];
        m.b1  = (const float*)d_in[b + 4]; m.w1  = prep(b + 5, K1, Kp1);
        m.b2  = (const float*)d_in[b + 6]; m.w2  = prep(b + 7, LAT, LAT);
        m.b3  = (const float*)d_in[b + 8]; m.w3  = prep(b + 9, LAT, LAT);
        return m;
    };

    MLPDev mNodeEnc = mkmlp(IDX_NODE_ENC, 145, 160);
    MLPDev mEdgeEnc = mkmlp(IDX_EDGE_ENC, 4, 32);
    MLPDev mEdge[10], mNode[10];
    for (int i = 0; i < 10; ++i) {
        mEdge[i] = mkmlp(PB + i * 20,      384, 384);
        mNode[i] = mkmlp(PB + i * 20 + 10, 256, 256);
    }

    const int nNodeBlk = (N + 15) / 16;
    const int nEdgeBlk = (E + 15) / 16;

    k_style<<<1, TPB, 0, stream>>>(features,
                                   (const float*)d_in[IDX_P1W], (const float*)d_in[IDX_P1B],
                                   (const float*)d_in[IDX_P2W], (const float*)d_in[IDX_P2B], style);
    k_node_encode<<<nNodeBlk, TPB, 0, stream>>>(pos, style, smpl, mat, batch, mNodeEnc, x, xb, N);
    k_edge_encode<<<nEdgeBlk, TPB, 0, stream>>>(edge_attr, mEdgeEnc, ea, E);

    for (int i = 0; i < 10; ++i) {
        hipMemsetAsync(msgs, 0, (size_t)N * LAT * 4, stream);
        k_edge_block<<<nEdgeBlk, TPB, 0, stream>>>(xb, edge_index, E, mEdge[i], ea, msgs);
        k_node_block<<<nNodeBlk, TPB, 0, stream>>>(msgs, mNode[i], x, xb, N);
    }
    k_decode<<<(N + TPB - 1) / TPB, TPB, 0, stream>>>(x, (const float*)d_in[IDX_DEC_W],
                                                      (const float*)d_in[IDX_DEC_B],
                                                      (float*)d_out, N);
}
